// GNNRetrievalModel_55929064128643
// MI455X (gfx1250) — matrix operand order repaired and checked by reading, not compiled, not run on hardware
//
#include <hip/hip_runtime.h>

// ---------------------------------------------------------------------------
// GNN retrieval model for MI455X (gfx1250, wave32, bf16 WMMA, LDS-free GEMM)
// ---------------------------------------------------------------------------

typedef __attribute__((ext_vector_type(16))) __bf16 bf16x16;
typedef __attribute__((ext_vector_type(8)))  float  v8f;

union Frag {
    bf16x16 v;
    uint4   q[2];
};

static __device__ __forceinline__ unsigned short f2bf(float x) {
    // round-to-nearest-even float -> bf16 bits
    unsigned int u = __float_as_uint(x);
    return (unsigned short)((u + 0x7FFFu + ((u >> 16) & 1u)) >> 16);
}

static __device__ __forceinline__ unsigned int pack2(float a, float b) {
    return (unsigned int)f2bf(a) | ((unsigned int)f2bf(b) << 16);
}

// ---------------------------------------------------------------------------
// small utility kernels
// ---------------------------------------------------------------------------
__global__ void zero_k(float* __restrict__ p, int n) {
    int i = blockIdx.x * blockDim.x + threadIdx.x;
    if (i < n) p[i] = 0.0f;
}

__global__ void deg_count(const int* __restrict__ dst, const int* __restrict__ et,
                          float* __restrict__ deg, int E, int R) {
    int e = blockIdx.x * blockDim.x + threadIdx.x;
    if (e < E) atomicAdd(&deg[dst[e] * R + et[e]], 1.0f);
}

__global__ void recip_k(float* __restrict__ p, int n) {
    int i = blockIdx.x * blockDim.x + threadIdx.x;
    if (i < n) p[i] = 1.0f / fmaxf(p[i], 1.0f);
}

// 128 threads per edge, 4 floats per thread, f32 atomics resolve in L2
__global__ void scatter_add(const int* __restrict__ src, const int* __restrict__ dst,
                            const int* __restrict__ et, const float* __restrict__ x,
                            float* __restrict__ agg, int E, int R) {
    int e = blockIdx.x * 2 + (threadIdx.x >> 7);
    if (e >= E) return;
    int f   = (threadIdx.x & 127) * 4;
    int s   = src[e];
    int seg = dst[e] * R + et[e];
    float4 v = *(const float4*)(x + s * 512 + f);
    float* a = agg + seg * 512 + f;
    atomicAdd(a + 0, v.x);
    atomicAdd(a + 1, v.y);
    atomicAdd(a + 2, v.z);
    atomicAdd(a + 3, v.w);
}

// ---------------------------------------------------------------------------
// A-operand prep: Abf[m][k] = bf16( concat(agg[m]*norm, root[m]) ), coalesced.
// One block per row m; thread handles 8 consecutive k (one b128 store).
// blockDim = Ktot/8 (256 for layers, 192 for context).  Krel = Ktot - 512.
// ---------------------------------------------------------------------------
__global__ void aprep(const float* __restrict__ agg, const float* __restrict__ norm,
                      const float* __restrict__ root, unsigned short* __restrict__ Abf,
                      int Ktot) {
    int m    = blockIdx.x;
    int k8   = threadIdx.x * 8;
    int Krel = Ktot - 512;
    int R    = Krel >> 9;
    float4 a, b;
    if (k8 < Krel) {
        float nm = norm[m * R + (k8 >> 9)];
        const float4* p = (const float4*)(agg + (size_t)m * Krel + k8);
        a = p[0]; b = p[1];
        a.x *= nm; a.y *= nm; a.z *= nm; a.w *= nm;
        b.x *= nm; b.y *= nm; b.z *= nm; b.w *= nm;
    } else {
        const float4* p = (const float4*)(root + (size_t)m * 512 + (k8 - Krel));
        a = p[0]; b = p[1];
    }
    uint4 q;
    q.x = pack2(a.x, a.y);
    q.y = pack2(a.z, a.w);
    q.z = pack2(b.x, b.y);
    q.w = pack2(b.z, b.w);
    *(uint4*)(Abf + (size_t)m * Ktot + k8) = q;
}

// ---------------------------------------------------------------------------
// W prep: WbfT[n][k] = bf16( W[k][n] ), W = [Wrel (Krel x 512) | Wroot (512 x 512)]
// grid (Ktot, 2), block 256; coalesced reads along n, scattered u16 writes (2MB once).
// ---------------------------------------------------------------------------
__global__ void wprep(const float* __restrict__ Wrel, const float* __restrict__ Wroot,
                      unsigned short* __restrict__ WbfT, int Ktot) {
    int k    = blockIdx.x;
    int n    = blockIdx.y * 256 + threadIdx.x;
    int Krel = Ktot - 512;
    const float* W = (k < Krel) ? (Wrel + (size_t)k * 512)
                                : (Wroot + (size_t)(k - Krel) * 512);
    WbfT[(size_t)n * Ktot + k] = f2bf(W[n]);
}

// ---------------------------------------------------------------------------
// bf16 WMMA GEMM, no LDS, no barriers:
//   out[m, n] = sum_k Abf[m,k] * Bbf[n,k]  (+ bias[n % 512] if bias != null)
// A: [M][Ktot] bf16 row-major.  B: [Nb][Ktot] bf16 row-major (column operand,
// already "transposed", so fragment loads are contiguous b128 pairs).
// Block = 256 thr = 8 waves: M-tile 16 x N-strip 512 (wave w owns cols w*64..+63,
// 4 sub-tiles of 16).  grid = (ceil(N/512), M/16).  M must be a multiple of 16.
// Per K-step/wave: 2 A b128 loads + 8 B b128 loads + 4 v_wmma_f32_16x16x32_bf16.
// ---------------------------------------------------------------------------
__global__ void __launch_bounds__(256)
gemm_bf16(const unsigned short* __restrict__ Abf, const unsigned short* __restrict__ Bbf,
          const float* __restrict__ bias, float* __restrict__ out,
          int Nb, int Ktot, int ldOut) {
    const int tid    = threadIdx.x;
    const int lane   = tid & 31;
    const int wave   = tid >> 5;
    const int kh     = lane >> 4;   // lane half selects K sub-range
    const int ln     = lane & 15;
    const int mBase  = blockIdx.y * 16;
    const int nBlock = blockIdx.x * 512;

    // A fragment row (ISA bf16 16x32 layout: both lane halves hold M = ln)
    const unsigned short* arow = Abf + (size_t)(mBase + ln) * Ktot;

    // B fragment rows; clamp out-of-range rows to 0 — their products land only
    // in columns the guarded store below never writes.
    const unsigned short* brow[4];
    int colg[4];
    for (int t = 0; t < 4; ++t) {
        int gp   = nBlock + wave * 64 + t * 16 + ln;
        colg[t]  = gp;
        int gpc  = (gp < Nb) ? gp : 0;
        brow[t]  = Bbf + (size_t)gpc * Ktot;
    }

    v8f acc[4] = {};

#pragma unroll 2
    for (int k0 = 0; k0 < Ktot; k0 += 32) {
        Frag af;
        af.q[0] = *(const uint4*)(arow + k0 + kh * 16);       // K = kh*16 + 0..7
        af.q[1] = *(const uint4*)(arow + k0 + kh * 16 + 8);   // K = kh*16 + 8..15

        for (int t = 0; t < 4; ++t) {
            Frag bf_;
            const uint4* pb = (const uint4*)(brow[t] + k0 + kh * 16); // K = kh*16+0..15
            bf_.q[0] = pb[0];
            bf_.q[1] = pb[1];
            acc[t] = __builtin_amdgcn_wmma_f32_16x16x32_bf16(
                false, af.v, false, bf_.v, (short)0, acc[t], false, false);
        }
    }

    for (int t = 0; t < 4; ++t) {
        if (colg[t] >= Nb) continue;
        float bb = bias ? bias[wave * 64 + t * 16 + ln] : 0.0f;
        for (int v = 0; v < 8; ++v) {
            int row = mBase + v + kh * 8;   // C/D layout: lane half adds 8 rows
            out[(size_t)row * ldOut + colg[t]] = acc[t][v] + bb;
        }
    }
}

// ---------------------------------------------------------------------------
// relu -> residual -> layernorm, one wave per row (H = 512), wave32 shuffles.
// Optionally emits a bf16 copy (feeds the scoring GEMM directly).
// ---------------------------------------------------------------------------
__global__ void __launch_bounds__(256)
relu_res_ln(const float* __restrict__ h, const float* __restrict__ res,
            const float* __restrict__ g, const float* __restrict__ b,
            float* __restrict__ out, unsigned short* __restrict__ outbf, int M) {
    int row = blockIdx.x * 8 + (threadIdx.x >> 5);
    if (row >= M) return;
    int lane = threadIdx.x & 31;
    float y[16];
    float s = 0.0f, sq = 0.0f;
    for (int j = 0; j < 16; ++j) {
        int e = lane + j * 32;
        float v = fmaxf(h[(size_t)row * 512 + e], 0.0f) + res[(size_t)row * 512 + e];
        y[j] = v;
        s  += v;
        sq += v * v;
    }
    for (int o = 16; o > 0; o >>= 1) {
        s  += __shfl_xor(s, o, 32);
        sq += __shfl_xor(sq, o, 32);
    }
    float mean = s * (1.0f / 512.0f);
    float var  = sq * (1.0f / 512.0f) - mean * mean;
    float inv  = rsqrtf(var + 1e-5f);
    for (int j = 0; j < 16; ++j) {
        int e = lane + j * 32;
        float r = (y[j] - mean) * inv * g[e] + b[e];
        out[(size_t)row * 512 + e] = r;
        if (outbf) outbf[(size_t)row * 512 + e] = f2bf(r);
    }
}

// ---------------------------------------------------------------------------
// launch
// ---------------------------------------------------------------------------
extern "C" void kernel_launch(void* const* d_in, const int* in_sizes, int n_in,
                              void* d_out, int out_size, void* d_ws, size_t ws_size,
                              hipStream_t stream) {
    const float* ctx    = (const float*)d_in[0];
    const float* prem   = (const float*)d_in[1];
    const int*   pei    = (const int*)d_in[2];
    const int*   pet    = (const int*)d_in[3];
    const int*   csrc   = (const int*)d_in[4];
    const int*   cdst   = (const int*)d_in[5];
    const int*   cet    = (const int*)d_in[6];
    const float* W1r    = (const float*)d_in[7];
    const float* W1root = (const float*)d_in[8];
    const float* b1     = (const float*)d_in[9];
    const float* g1     = (const float*)d_in[10];
    const float* bb1    = (const float*)d_in[11];
    const float* W2r    = (const float*)d_in[12];
    const float* W2root = (const float*)d_in[13];
    const float* b2     = (const float*)d_in[14];
    const float* g2     = (const float*)d_in[15];
    const float* bb2    = (const float*)d_in[16];
    const float* Wcr    = (const float*)d_in[17];
    const float* Wcroot = (const float*)d_in[18];
    const float* bc     = (const float*)d_in[19];
    const float* gc     = (const float*)d_in[20];
    const float* bbc    = (const float*)d_in[21];

    const int N_P = in_sizes[1] / 512;   // 20000 (multiple of 16)
    const int N_C = in_sizes[0] / 512;   // 512
    const int E_P = in_sizes[3];         // 320000
    const int E_C = in_sizes[4];         // 262144

    const int* psrc = pei;
    const int* pdst = pei + E_P;

    // ---- workspace layout -------------------------------------------------
    float* ws    = (float*)d_ws;
    float* agg   = ws;                               // 30,720,000 f32 (N_P*3*512)
    float* normP = agg + (size_t)N_P * 3 * 512;      //     60,000
    float* normC = normP + (size_t)N_P * 3;          //      1,024 (padded)
    float* hbuf  = normC + 2048;                     // 10,240,000
    float* x1    = hbuf + (size_t)N_P * 512;         // 10,240,000
    float* pf    = x1 + (size_t)N_P * 512;           // 10,240,000
    float* cf    = pf + (size_t)N_P * 512;           //    262,144
    unsigned short* Abf  = (unsigned short*)(cf + (size_t)N_C * 512); // 40,960,000 bf16
    unsigned short* WbfT = Abf + (size_t)N_P * 2048;                  //  1,048,576 bf16
    unsigned short* pfbf = WbfT + (size_t)512 * 2048;                 // 10,240,000 bf16
    unsigned short* cfbf = pfbf + (size_t)N_P * 512;                  //    262,144 bf16
    float* scores = (float*)d_out;                                    // [N_C, N_P]

    const int aggP = N_P * 3 * 512;
    const int aggC = N_C * 2 * 512;
    const int KL   = 2048;               // (3+1)*512 premise layers
    const int KC   = 1536;               // (2+1)*512 context layer

    // ---- premise-graph normalization (shared by both layers)
    zero_k<<<(N_P * 3 + 255) / 256, 256, 0, stream>>>(normP, N_P * 3);
    deg_count<<<(E_P + 255) / 256, 256, 0, stream>>>(pdst, pet, normP, E_P, 3);
    recip_k<<<(N_P * 3 + 255) / 256, 256, 0, stream>>>(normP, N_P * 3);

    // ---- layer 1
    zero_k<<<(aggP + 255) / 256, 256, 0, stream>>>(agg, aggP);
    scatter_add<<<(E_P + 1) / 2, 256, 0, stream>>>(psrc, pdst, pet, prem, agg, E_P, 3);
    aprep<<<N_P, KL / 8, 0, stream>>>(agg, normP, prem, Abf, KL);
    wprep<<<dim3(KL, 2), 256, 0, stream>>>(W1r, W1root, WbfT, KL);
    gemm_bf16<<<dim3(1, N_P / 16), 256, 0, stream>>>(Abf, WbfT, b1, hbuf, 512, KL, 512);
    relu_res_ln<<<(N_P + 7) / 8, 256, 0, stream>>>(hbuf, prem, g1, bb1, x1, nullptr, N_P);

    // ---- layer 2
    zero_k<<<(aggP + 255) / 256, 256, 0, stream>>>(agg, aggP);
    scatter_add<<<(E_P + 1) / 2, 256, 0, stream>>>(psrc, pdst, pet, x1, agg, E_P, 3);
    aprep<<<N_P, KL / 8, 0, stream>>>(agg, normP, x1, Abf, KL);
    wprep<<<dim3(KL, 2), 256, 0, stream>>>(W2r, W2root, WbfT, KL);
    gemm_bf16<<<dim3(1, N_P / 16), 256, 0, stream>>>(Abf, WbfT, b2, hbuf, 512, KL, 512);
    relu_res_ln<<<(N_P + 7) / 8, 256, 0, stream>>>(hbuf, x1, g2, bb2, pf, pfbf, N_P);

    // ---- context layer (premise -> context bipartite)
    zero_k<<<(N_C * 2 + 255) / 256, 256, 0, stream>>>(normC, N_C * 2);
    deg_count<<<(E_C + 255) / 256, 256, 0, stream>>>(cdst, cet, normC, E_C, 2);
    recip_k<<<(N_C * 2 + 255) / 256, 256, 0, stream>>>(normC, N_C * 2);

    zero_k<<<(aggC + 255) / 256, 256, 0, stream>>>(agg, aggC);
    scatter_add<<<(E_C + 1) / 2, 256, 0, stream>>>(csrc, cdst, cet, pf, agg, E_C, 2);
    aprep<<<N_C, KC / 8, 0, stream>>>(agg, normC, ctx, Abf, KC);
    wprep<<<dim3(KC, 2), 256, 0, stream>>>(Wcr, Wcroot, WbfT, KC);
    gemm_bf16<<<dim3(1, N_C / 16), 256, 0, stream>>>(Abf, WbfT, bc, hbuf, 512, KC, 512);
    relu_res_ln<<<(N_C + 7) / 8, 256, 0, stream>>>(hbuf, ctx, gc, bbc, cf, cfbf, N_C);

    // ---- retrieval scores: c_feats @ p_feats^T  (both already bf16)
    gemm_bf16<<<dim3((N_P + 511) / 512, N_C / 16), 256, 0, stream>>>(
        cfbf, pfbf, nullptr, scores, N_P, 512, N_P);
}
